// ConsistentSelfAttentionTile_62277025792229
// MI455X (gfx1250) — matrix operand order repaired and checked by reading, not compiled, not run on hardware
//
#include <hip/hip_runtime.h>
#include <hip/hip_bf16.h>

typedef __bf16 bf16_t;
typedef __attribute__((ext_vector_type(16))) __bf16 v16bf;
typedef __attribute__((ext_vector_type(8)))  float  v8f;
typedef unsigned int v4u __attribute__((ext_vector_type(4)));
typedef int          v8i __attribute__((ext_vector_type(8)));
typedef int          v4i __attribute__((ext_vector_type(4)));

#define B_   2
#define N_   512
#define C_   512
#define W_   64
#define T_   449
#define S_   288
#define VH_  256          // V staged in two column halves of 256
#define VP_  264          // LDS pitch for V half (256 + 8 pad elements per row)

__device__ __forceinline__ v8f wmma_bf16(v16bf a, v16bf b, v8f c) {
  // D = A(16x32 bf16) * B(32x16 bf16) + C(16x16 f32)
  return __builtin_amdgcn_wmma_f32_16x16x32_bf16(false, a, false, b, (short)0, c,
                                                 false, false);
}

// ---- Tensor Data Mover: 2D tile Global -> LDS with per-interval LDS padding ----
// ISA 8.3-8.5 descriptor. dims/strides in data_size units (4B here). Pads
// pad_amount dwords into LDS after every pad_interval dwords (load only).
__device__ __forceinline__ void tdm_load_2d(unsigned lds_addr, const void* gaddr,
                                            unsigned tile_w_dw, unsigned tile_h,
                                            unsigned row_stride_dw,
                                            unsigned pad_interval_code,
                                            unsigned pad_amount_code) {
  unsigned long long ga = (unsigned long long)(uintptr_t)gaddr;
  v4u g0;
  g0[0] = 1u;                                            // count=1, no gather
  g0[1] = lds_addr;                                      // LDS byte address
  g0[2] = (unsigned)ga;                                  // global_addr[31:0]
  g0[3] = (unsigned)((ga >> 32) & 0x1FFFFFFu) | (2u << 30); // [56:32] + type=2
  unsigned tdim0 = 0x40000000u, tdim1 = 0x40000000u;     // huge: no OOB clip
  v8i g1;
  g1[0] = (int)((2u << 16)                               // data_size = 4 bytes
              | (1u << 20)                               // pad_enable
              | (pad_interval_code << 22)
              | (pad_amount_code << 25));
  g1[1] = (int)((tdim0 & 0xFFFFu) << 16);                // bar_addr=0 | tdim0.lo
  g1[2] = (int)(((tdim0 >> 16) & 0xFFFFu) | ((tdim1 & 0xFFFFu) << 16));
  g1[3] = (int)(((tdim1 >> 16) & 0xFFFFu) | (tile_w_dw << 16)); // tile_dim0
  g1[4] = (int)(tile_h & 0xFFFFu);                       // tile_dim1; dim2=0
  g1[5] = (int)row_stride_dw;                            // dim0_stride[31:0]
  g1[6] = 0;                                             // stride hi, dim1_stride
  g1[7] = 0;
  v4i z4 = {0, 0, 0, 0};                                 // 2D: groups 2/3 unused
  v8i z8 = {0, 0, 0, 0, 0, 0, 0, 0};
  __builtin_amdgcn_tensor_load_to_lds(g0, g1, z4, z4, z8, 0);
}

// ---- WMMA fragment loaders (wave32; contiguous k = 16h + s per lane half, same for A and B) ----
// A operand 16x32: lane l -> m=l%16, h=l/16; holds K in [16h,16h+16)
__device__ __forceinline__ v16bf ld_fragA(const bf16_t* src, int ld, int row0,
                                          int k0, int lane) {
  const bf16_t* p = src + (size_t)(row0 + (lane & 15)) * ld + k0 + ((lane >> 4) << 4);
  v16bf a;
#pragma unroll
  for (int i = 0; i < 16; ++i) a[i] = p[i];
  return a;
}
__device__ __forceinline__ v16bf ld_fragA_f32(const float* src, int ld, int row0,
                                              int k0, int lane) {
  const float* p = src + (size_t)(row0 + (lane & 15)) * ld + k0 + ((lane >> 4) << 4);
  v16bf a;
#pragma unroll
  for (int i = 0; i < 16; ++i) a[i] = (bf16_t)p[i];
  return a;
}
// B operand 32x16 built from B^T stored row-major [N][K] (B[k][n] = srcT[n][k]).
// Lane l -> n=l%16, h=l/16; holds K in [16h,16h+16)  => 16 contiguous elements.
__device__ __forceinline__ v16bf ld_fragBT(const bf16_t* srcT, int ld, int n0,
                                           int k0, int lane) {
  const bf16_t* p = srcT + (size_t)(n0 + (lane & 15)) * ld + k0 + ((lane >> 4) << 4);
  v16bf b;
#pragma unroll
  for (int i = 0; i < 16; ++i) b[i] = p[i];
  return b;
}
__device__ __forceinline__ v16bf ld_fragBT_f32(const float* srcT, int ld, int n0,
                                               int k0, int lane) {
  const float* p = srcT + (size_t)(n0 + (lane & 15)) * ld + k0 + ((lane >> 4) << 4);
  v16bf b;
#pragma unroll
  for (int i = 0; i < 16; ++i) b[i] = (bf16_t)p[i];
  return b;
}
// B operand 32x16 from row-major B [K][N] (per-lane gather; use on LDS).
__device__ __forceinline__ v16bf ld_fragB(const bf16_t* src, int ld, int k0,
                                          int n0, int lane) {
  const bf16_t* p = src + (size_t)(k0 + ((lane >> 4) << 4)) * ld + n0 + (lane & 15);
  v16bf b;
#pragma unroll
  for (int i = 0; i < 16; ++i) b[i] = p[(size_t)i * ld];
  return b;
}

// ---- Kernel 1: Q/K/V projections for all 1024 tokens, fp32 in -> bf16 out ----
// Y[m][n] = sum_k X[m][k] * W[n][k] + bias[n]   (W row-major [out][in] == B^T)
__global__ __launch_bounds__(256) void proj_qkv_kernel(
    const float* __restrict__ X,
    const float* __restrict__ Wq, const float* __restrict__ bq,
    const float* __restrict__ Wk, const float* __restrict__ bk,
    const float* __restrict__ Wv, const float* __restrict__ bv,
    bf16_t* __restrict__ Qb, bf16_t* __restrict__ Kb, bf16_t* __restrict__ Vb) {
  int lane = threadIdx.x & 31;
  int g    = blockIdx.x * 8 + (threadIdx.x >> 5);   // global wave id, 0..6143
  int proj = g >> 11;                               // 0..2  (2048 tiles each)
  int rem  = g & 2047;
  int mt   = rem >> 5;                              // 0..63  (1024/16 row tiles)
  int nt   = rem & 31;                              // 0..31  (512/16 col tiles)
  const float* Wsel = proj == 0 ? Wq : (proj == 1 ? Wk : Wv);
  const float* bsel = proj == 0 ? bq : (proj == 1 ? bk : bv);
  bf16_t*      Ysel = proj == 0 ? Qb : (proj == 1 ? Kb : Vb);

  v8f acc = {};
#pragma unroll 4
  for (int kk = 0; kk < C_; kk += 32) {
    v16bf a = ld_fragA_f32(X, C_, mt * 16, kk, lane);
    v16bf b = ld_fragBT_f32(Wsel, C_, nt * 16, kk, lane);
    acc = wmma_bf16(a, b, acc);
  }
  int n     = nt * 16 + (lane & 15);
  float bia = bsel[n];
  int mrow0 = mt * 16 + ((lane >> 4) << 3);
#pragma unroll
  for (int r = 0; r < 8; ++r)
    Ysel[(size_t)(mrow0 + r) * C_ + n] = (bf16_t)(acc[r] + bia);
}

// ---- Kernel 2: per-tile key multiplicities  c_j = 1 + #{s : idx[t][s]%64 == j} ----
__global__ __launch_bounds__(256) void counts_kernel(const int* __restrict__ sidx,
                                                     float* __restrict__ cnts) {
  __shared__ int sh[S_];
  int t = blockIdx.x;
  for (int s = threadIdx.x; s < S_; s += 256) sh[s] = sidx[t * S_ + s];
  __syncthreads();
  if (threadIdx.x < W_) {
    int j = threadIdx.x;
    int c = 1;                       // the appended tile token itself
#pragma unroll 8
    for (int s = 0; s < S_; ++s) c += ((sh[s] & (W_ - 1)) == j) ? 1 : 0;
    cnts[t * W_ + j] = (float)c;
  }
}

// ---- Kernel 3: per-(tile,batch) attention with multiplicity softmax ----
__global__ __launch_bounds__(256) void attn_tile_kernel(
    const bf16_t* __restrict__ Qb, const bf16_t* __restrict__ Kb,
    const bf16_t* __restrict__ Vb, const float* __restrict__ cnts,
    float* __restrict__ Obuf) {
  __shared__ float  Ls[W_][W_];                        // 16 KB
  __shared__ alignas(16) bf16_t Ps[W_][W_];            //  8 KB
  __shared__ alignas(16) bf16_t Vs[W_ * VP_];          // 33 KB (one V column-half)
  __shared__ float  cs[W_];
  int t = blockIdx.x, b = blockIdx.y;
  int tid  = threadIdx.x;
  int lane = tid & 31;
  int w    = tid >> 5;                                 // wave 0..7
  size_t row0 = (size_t)b * N_ + t;                    // first token row of tile

  if (tid < W_) cs[tid] = cnts[t * W_ + tid];

  const bf16_t* Qt = Qb + row0 * C_;
  const bf16_t* Kt = Kb + row0 * C_;
  const bf16_t* Vt = Vb + row0 * C_;

  // Kick off async TDM stage of V column-half 0 (64 rows x 512B) into LDS.
  // Row stride 256 dw; +4 dw LDS padding after every 128 dw -> pitch = VP_.
  if (w == 0)
    tdm_load_2d((unsigned)(uintptr_t)&Vs[0], Vt, VH_ / 2, W_, C_ / 2, 6, 3);

  // L = Q * K^T  (64x64 f32): wave w owns tiles (w>>1, (w&1)*2) and (w>>1, (w&1)*2+1)
  {
    int ti  = w >> 1;
    int tj0 = (w & 1) * 2;
    v8f acc0 = {}, acc1 = {};
    for (int kk = 0; kk < C_; kk += 32) {
      v16bf a  = ld_fragA(Qt, C_, ti * 16, kk, lane);              // shared A
      v16bf b0 = ld_fragBT(Kt, C_, tj0 * 16, kk, lane);            // K rows = B^T
      v16bf b1 = ld_fragBT(Kt, C_, tj0 * 16 + 16, kk, lane);
      acc0 = wmma_bf16(a, b0, acc0);
      acc1 = wmma_bf16(a, b1, acc1);
    }
    int nn  = lane & 15;
    int mm0 = ti * 16 + ((lane >> 4) << 3);
#pragma unroll
    for (int r = 0; r < 8; ++r) {
      Ls[mm0 + r][tj0 * 16 + nn]      = acc0[r];
      Ls[mm0 + r][tj0 * 16 + 16 + nn] = acc1[r];
    }
  }
  __syncthreads();

  // Softmax over 64 unique keys with multiplicities cs[j] (== softmax over 352 dups)
  if (tid < W_) {
    int row = tid;
    float m = -1e30f;
    for (int j = 0; j < W_; ++j) m = fmaxf(m, Ls[row][j]);
    float z = 0.f;
    for (int j = 0; j < W_; ++j) z += cs[j] * __expf(Ls[row][j] - m);
    float inv = 1.0f / z;
    for (int j = 0; j < W_; ++j)
      Ps[row][j] = (bf16_t)(cs[j] * __expf(Ls[row][j] - m) * inv);
  }
  if (w == 0) __builtin_amdgcn_s_wait_tensorcnt(0);    // V half-0 landed in LDS
  __syncthreads();                                     // Ps + Vs visible to all

  // O = P(64x64) * V(64x512): wave w owns row tile (w&3), 8 col tiles per V-half.
  // A fragments (P rows) loaded once per wave, reused across all col tiles.
  int ti3 = w & 3;
  int jb  = (w >> 2) * 8;                              // col-tile base within half
  v16bf a0 = ld_fragA(&Ps[0][0], W_, ti3 * 16, 0, lane);
  v16bf a1 = ld_fragA(&Ps[0][0], W_, ti3 * 16, 32, lane);
  int mm0 = ti3 * 16 + ((lane >> 4) << 3);
  int nn  = lane & 15;
  float* Ot = Obuf + ((size_t)t * B_ + b) * (size_t)(W_ * C_);

#pragma unroll
  for (int half = 0; half < 2; ++half) {
    if (half == 1) {
      __syncthreads();                                 // everyone done with half 0
      if (w == 0) {                                    // TDM stage half 1
        tdm_load_2d((unsigned)(uintptr_t)&Vs[0],
                    (const void*)((const char*)Vt + VH_ * 2 /*bytes*/),
                    VH_ / 2, W_, C_ / 2, 6, 3);
        __builtin_amdgcn_s_wait_tensorcnt(0);
      }
      __syncthreads();
    }
#pragma unroll 2
    for (int j = 0; j < 8; ++j) {
      int tj = jb + j;                                 // local col tile in half
      v8f acc = {};
      acc = wmma_bf16(a0, ld_fragB(Vs, VP_, 0, tj * 16, lane), acc);
      acc = wmma_bf16(a1, ld_fragB(Vs, VP_, 32, tj * 16, lane), acc);
      int col = half * VH_ + tj * 16 + nn;
#pragma unroll
      for (int r = 0; r < 8; ++r)
        Ot[(size_t)(mm0 + r) * C_ + col] = acc[r];
    }
  }
}

// ---- Kernel 4: deterministic overlap reduce + count division ----
__global__ __launch_bounds__(256) void reduce_kernel(const float* __restrict__ Obuf,
                                                     float* __restrict__ out) {
  int gid = blockIdx.x * 256 + threadIdx.x;       // 0 .. B*N*C-1
  int c = gid & (C_ - 1);
  int n = (gid >> 9) & (N_ - 1);
  int b = gid >> 18;
  int tlo = n - (W_ - 1); if (tlo < 0) tlo = 0;
  int thi = n;            if (thi > T_ - 1) thi = T_ - 1;
  float acc = 0.f;
  for (int t = tlo; t <= thi; ++t) {
    int r = n - t;
    acc += Obuf[(((size_t)t * B_ + b) * W_ + r) * C_ + c];
  }
  out[gid] = acc / (float)(thi - tlo + 1);
}

extern "C" void kernel_launch(void* const* d_in, const int* in_sizes, int n_in,
                              void* d_out, int out_size, void* d_ws, size_t ws_size,
                              hipStream_t stream) {
  const float* X  = (const float*)d_in[0];
  const float* Wq = (const float*)d_in[1];
  const float* bq = (const float*)d_in[2];
  const float* Wk = (const float*)d_in[3];
  const float* bk = (const float*)d_in[4];
  const float* Wv = (const float*)d_in[5];
  const float* bv = (const float*)d_in[6];
  const int* sidx = (const int*)d_in[7];
  float* out = (float*)d_out;

  // Workspace layout (~121 MB):
  //   [0,1MB)   Qb  bf16 [1024][512]
  //   [1,2MB)   Kb  bf16 [1024][512]
  //   [2,3MB)   Vb  bf16 [1024][512]
  //   [3,4MB)   cnts f32 [449][64]
  //   [4MB,..)  Obuf f32 [449][2][64][512]
  char* ws = (char*)d_ws;
  bf16_t* Qb   = (bf16_t*)(ws);
  bf16_t* Kb   = (bf16_t*)(ws + (size_t)(1u << 20));
  bf16_t* Vb   = (bf16_t*)(ws + (size_t)2 * (1u << 20));
  float*  cnts = (float*) (ws + (size_t)3 * (1u << 20));
  float*  Obuf = (float*) (ws + (size_t)4 * (1u << 20));

  proj_qkv_kernel<<<768, 256, 0, stream>>>(X, Wq, bq, Wk, bk, Wv, bv, Qb, Kb, Vb);
  counts_kernel<<<T_, 256, 0, stream>>>(sidx, cnts);
  attn_tile_kernel<<<dim3(T_, B_), 256, 0, stream>>>(Qb, Kb, Vb, cnts, Obuf);
  reduce_kernel<<<(B_ * N_ * C_) / 256, 256, 0, stream>>>(Obuf, out);
}
